// DAMT_12197707120749
// MI455X (gfx1250) — matrix operand order repaired and checked by reading, not compiled, not run on hardware
//
#include <hip/hip_runtime.h>

// ---------------------------------------------------------------------------
// CDNA5 (gfx1250) fused dual cross-attention (DAMT) using bf16 WMMA.
// Wave32 only. Matrix math via v_wmma_f32_16x16x32_bf16; GEMM B-operand
// staging via GLOBAL_LOAD_ASYNC_TO_LDS_B128 (ASYNCcnt), double-buffered and
// software-pipelined so the async transfer overlaps the WMMA stream.
// ~250 MB workspace.
// ---------------------------------------------------------------------------

#define USE_ASYNC_LDS 1

typedef __attribute__((ext_vector_type(16))) __bf16 v16bf;
typedef __attribute__((ext_vector_type(8)))  float  v8f;

#define WMMA_BF16(A, B, C) \
  __builtin_amdgcn_wmma_f32_16x16x32_bf16(false, (A), false, (B), (short)0, (C), false, false)

__device__ __forceinline__ unsigned short f2bf(float f) {
  unsigned int u = __float_as_uint(f);
  u += 0x7FFFu + ((u >> 16) & 1u);            // round-to-nearest-even
  return (unsigned short)(u >> 16);
}

union V16U { uint4 q[2]; v16bf v; };

// A operand, 16x32 bf16 (ISA 7.12.2): lane m = lane&15, half kh = lane>>4.
// VGPR0-3 hold K = kh*8 + [0..8), VGPR4-7 hold K = 16 + kh*8 + [0..8).
__device__ __forceinline__ v16bf load_A(const unsigned short* rowp, int k0, int kh) {
  V16U u;
  u.q[0] = *(const uint4*)(rowp + k0 + kh * 8);
  u.q[1] = *(const uint4*)(rowp + k0 + 16 + kh * 8);
  return u.v;
}

// B operand, 32x16 bf16: lane n = lane&15, half kh = lane>>4.
// VGPR0-3 hold K = kh*8 + [0..8), VGPR4-7 hold K = 16 + kh*8 + [0..8).
__device__ __forceinline__ v16bf load_B(const unsigned short* rowp, int k0, int kh) {
  V16U u;
  u.q[0] = *(const uint4*)(rowp + k0 + kh * 8);
  u.q[1] = *(const uint4*)(rowp + k0 + 16 + kh * 8);
  return u.v;
}

#if USE_ASYNC_LDS
// Async global -> LDS copy, 16B per lane, tracked by ASYNCcnt.
__device__ __forceinline__ void async_ld_b128(unsigned lds_off, const void* gptr) {
  asm volatile("global_load_async_to_lds_b128 %0, %1, off"
               :: "v"(lds_off), "v"((unsigned long long)(uintptr_t)gptr)
               : "memory");
}
__device__ __forceinline__ void wait_asynccnt0() {
  asm volatile("s_wait_asynccnt 0x0" ::: "memory");
}
#endif

// ---------------------------------------------------------------------------
// fp32 -> bf16 conversion
// ---------------------------------------------------------------------------
__global__ __launch_bounds__(256) void cvt_bf16(const float* __restrict__ src,
                                                unsigned short* __restrict__ dst, int n) {
  int i = blockIdx.x * 256 + threadIdx.x;
  if (i < n) dst[i] = f2bf(src[i]);
}

// ---------------------------------------------------------------------------
// GEMM: Y[M,N] = X[M,K](bf16) * W[N,K]^T (bf16) + bias, M = 8192.
// Block: 256 threads = 8 waves arranged 2(M) x 4(N); wave tile = 32x64
// (2 M-tiles x 4 N-tiles). Block tile = 64 x 256. The 32x256 W tile for
// k-step i+1 is staged into the alternate LDS buffer by async-to-LDS loads
// issued before the step-i WMMAs; one s_wait_asynccnt + barrier per step.
// MODE 0: out bf16 row-major [M,N]
// MODE 1: out bf16 per-head transposed [b, h, d, s]  (N==1024, dh=256)
// MODE 2: out f32 = Y + resid  (row-major [M,N])
// ---------------------------------------------------------------------------
template <int MODE>
__global__ __launch_bounds__(256) void gemm_proj(
    const unsigned short* __restrict__ X, const unsigned short* __restrict__ W,
    const float* __restrict__ bias, const float* __restrict__ resid,
    unsigned short* __restrict__ out_bf, float* __restrict__ out_f, int N, int K) {
  __shared__ __align__(16) unsigned short Bt[2][256 * 32];  // 2 x 16KB [n_local][k]
  const int tid = threadIdx.x;
  const int wave = tid >> 5, lane = tid & 31;
  const int ln = lane & 15, kh = lane >> 4;
  const int wm = wave & 1, wn = wave >> 1;
  const int n0b = blockIdx.x * 256;            // block col base
  const int n0 = n0b + wn * 64;                // wave col base
  const int m0 = blockIdx.y * 64 + wm * 32;    // wave row base

  v8f acc[2][4];
#pragma unroll
  for (int mt = 0; mt < 2; mt++)
#pragma unroll
    for (int t = 0; t < 4; t++)
#pragma unroll
      for (int e = 0; e < 8; e++) acc[mt][t][e] = 0.f;

  const unsigned short* arow0 = X + (size_t)(m0 + ln) * K;
  const unsigned short* arow1 = arow0 + (size_t)16 * K;
  const unsigned short* wsrc = W + (size_t)(n0b + tid) * K;  // this thread's staged row
#if USE_ASYNC_LDS
  const unsigned ldsb0 = (unsigned)(uintptr_t)(Bt[0] + (size_t)tid * 32);
  const unsigned ldsb1 = (unsigned)(uintptr_t)(Bt[1] + (size_t)tid * 32);

  // Prologue: stage k-step 0 into buffer 0.
  async_ld_b128(ldsb0 + 0,  wsrc + 0);
  async_ld_b128(ldsb0 + 16, wsrc + 8);
  async_ld_b128(ldsb0 + 32, wsrc + 16);
  async_ld_b128(ldsb0 + 48, wsrc + 24);
  wait_asynccnt0();
  __syncthreads();

  for (int k0 = 0; k0 < K; k0 += 32) {
    const int buf = (k0 >> 5) & 1;
    if (k0 + 32 < K) {  // stage next k-step into the other buffer (overlapped)
      const unsigned dst = buf ? ldsb0 : ldsb1;
      const unsigned short* src = wsrc + k0 + 32;
      async_ld_b128(dst + 0,  src + 0);
      async_ld_b128(dst + 16, src + 8);
      async_ld_b128(dst + 32, src + 16);
      async_ld_b128(dst + 48, src + 24);
    }
    const unsigned short* bbase = Bt[buf];
    v16bf A0 = load_A(arow0, k0, kh);
    v16bf A1 = load_A(arow1, k0, kh);
#pragma unroll
    for (int t = 0; t < 4; t++) {
      v16bf Bv = load_B(bbase + (size_t)(wn * 64 + t * 16 + ln) * 32, 0, kh);
      acc[0][t] = WMMA_BF16(A0, Bv, acc[0][t]);
      acc[1][t] = WMMA_BF16(A1, Bv, acc[1][t]);
    }
    wait_asynccnt0();   // next buffer's stage complete (this wave)
    __syncthreads();    // all waves done reading buf / staging next
  }
#else
  for (int k0 = 0; k0 < K; k0 += 32) {
    __syncthreads();
    {
      uint4* dst = (uint4*)(Bt[0] + (size_t)tid * 32);
      const uint4* src = (const uint4*)(wsrc + k0);
      dst[0] = src[0]; dst[1] = src[1]; dst[2] = src[2]; dst[3] = src[3];
    }
    __syncthreads();
    v16bf A0 = load_A(arow0, k0, kh);
    v16bf A1 = load_A(arow1, k0, kh);
#pragma unroll
    for (int t = 0; t < 4; t++) {
      v16bf Bv = load_B(Bt[0] + (size_t)(wn * 64 + t * 16 + ln) * 32, 0, kh);
      acc[0][t] = WMMA_BF16(A0, Bv, acc[0][t]);
      acc[1][t] = WMMA_BF16(A1, Bv, acc[1][t]);
    }
  }
#endif

#pragma unroll
  for (int mt = 0; mt < 2; mt++) {
#pragma unroll
    for (int t = 0; t < 4; t++) {
      const int n = n0 + t * 16 + ln;
      const float bn = bias[n];
#pragma unroll
      for (int r = 0; r < 8; r++) {
        const int m = m0 + mt * 16 + r + kh * 8;
        float v = acc[mt][t][r] + bn;
        if (MODE == 0) {
          out_bf[(size_t)m * N + n] = f2bf(v);
        } else if (MODE == 1) {
          const int b = m >> 10, s = m & 1023, h = n >> 8, d = n & 255;
          out_bf[(((size_t)((b * 4 + h) * 256 + d)) << 10) + s] = f2bf(v);
        } else {
          const size_t idx = (size_t)m * N + n;
          out_f[idx] = v + resid[idx];
        }
      }
    }
  }
}

// ---------------------------------------------------------------------------
// Fused attention: one 16-row q-tile per wave, online softmax over 32-key
// steps, PV uses K=32 WMMA (two 16-col score tiles per step).
// Q,K: [8192, 2048] bf16 (4 heads x 512); Vt: [B*4*256, 1024] bf16 (d-major).
// ctx out: [8192, 1024] bf16 merged heads.
// ---------------------------------------------------------------------------
__global__ __launch_bounds__(128) void attn_fused(
    const unsigned short* __restrict__ Q, const unsigned short* __restrict__ Kc,
    const unsigned short* __restrict__ Vt, const float* __restrict__ mask,
    unsigned short* __restrict__ ctx) {
  __shared__ __align__(16) unsigned short Pl[4][16 * 32];  // 1KB per wave
  const int tid = threadIdx.x;
  const int wave = tid >> 5, lane = tid & 31;
  const int ln = lane & 15, kh = lane >> 4;
  const int wid = blockIdx.x * 4 + wave;
  const int qt = wid & 63, h = (wid >> 6) & 3, b = wid >> 8;
  const int q0 = qt * 16;

  const unsigned short* Qb = Q + (size_t)(b * 1024) * 2048 + h * 512;
  const unsigned short* Kb = Kc + (size_t)(b * 1024) * 2048 + h * 512;
  const unsigned short* Vb = Vt + (size_t)((b * 4 + h) * 256) * 1024;
  const float* mb = mask + b * 1024;
  const float scale = 0.04419417382415922f;  // 1/sqrt(512)

  v8f O[16];
#pragma unroll
  for (int i = 0; i < 16; i++)
#pragma unroll
    for (int e = 0; e < 8; e++) O[i][e] = 0.f;
  float mrun[8], lrun[8];
#pragma unroll
  for (int r = 0; r < 8; r++) { mrun[r] = -__builtin_inff(); lrun[r] = 0.f; }

  const unsigned short* qrow = Qb + (size_t)(q0 + ln) * 2048;
  unsigned short* pl = Pl[wave];

  for (int j0 = 0; j0 < 1024; j0 += 32) {
    v8f S0, S1;
#pragma unroll
    for (int e = 0; e < 8; e++) { S0[e] = 0.f; S1[e] = 0.f; }
    const unsigned short* k0row = Kb + (size_t)(j0 + ln) * 2048;
    const unsigned short* k1row = k0row + (size_t)16 * 2048;
    if (j0 + 32 < 1024) {  // prefetch next key tile's K rows (global_prefetch_b8)
      __builtin_prefetch(k0row + (size_t)32 * 2048, 0, 1);
      __builtin_prefetch(k1row + (size_t)32 * 2048, 0, 1);
      __builtin_prefetch(Vb + (size_t)ln * 16 * 1024 + j0 + 32, 0, 1);
    }
#pragma unroll 4
    for (int kk = 0; kk < 512; kk += 32) {
      v16bf A = load_A(qrow, kk, kh);
      S0 = WMMA_BF16(A, load_B(k0row, kk, kh), S0);
      S1 = WMMA_BF16(A, load_B(k1row, kk, kh), S1);
    }
    const float am0 = (1.f - mb[j0 + ln]) * -10000.f;
    const float am1 = (1.f - mb[j0 + 16 + ln]) * -10000.f;
    float p0[8], p1[8], alpha[8];
#pragma unroll
    for (int r = 0; r < 8; r++) {
      float s0 = S0[r] * scale + am0;
      float s1 = S1[r] * scale + am1;
      float mx = fmaxf(s0, s1);
      mx = fmaxf(mx, __shfl_xor(mx, 1, 32));
      mx = fmaxf(mx, __shfl_xor(mx, 2, 32));
      mx = fmaxf(mx, __shfl_xor(mx, 4, 32));
      mx = fmaxf(mx, __shfl_xor(mx, 8, 32));
      const float mn = fmaxf(mrun[r], mx);
      alpha[r] = __expf(mrun[r] - mn);
      mrun[r] = mn;
      const float e0 = __expf(s0 - mn), e1 = __expf(s1 - mn);
      float rs = e0 + e1;
      rs += __shfl_xor(rs, 1, 32);
      rs += __shfl_xor(rs, 2, 32);
      rs += __shfl_xor(rs, 4, 32);
      rs += __shfl_xor(rs, 8, 32);
      lrun[r] = lrun[r] * alpha[r] + rs;
      p0[r] = e0; p1[r] = e1;
    }
#pragma unroll
    for (int i = 0; i < 16; i++)
#pragma unroll
      for (int r = 0; r < 8; r++) O[i][r] *= alpha[r];

    // Relayout P (C layout) -> A layout via LDS.
#pragma unroll
    for (int r = 0; r < 8; r++) {
      const int row = r + kh * 8;
      pl[row * 32 + ln] = f2bf(p0[r]);
      pl[row * 32 + 16 + ln] = f2bf(p1[r]);
    }
    __syncthreads();
    v16bf Ap = load_A(pl + ln * 32, 0, kh);
    __syncthreads();
#pragma unroll
    for (int i = 0; i < 16; i++) {
      v16bf Bv = load_B(Vb + (size_t)(i * 16 + ln) * 1024, j0, kh);
      O[i] = WMMA_BF16(Ap, Bv, O[i]);
    }
  }

  float inv[8];
#pragma unroll
  for (int r = 0; r < 8; r++) inv[r] = (lrun[r] > 0.f) ? (1.f / lrun[r]) : 0.f;
#pragma unroll
  for (int i = 0; i < 16; i++) {
    const int n = h * 256 + i * 16 + ln;
#pragma unroll
    for (int r = 0; r < 8; r++) {
      const int m = b * 1024 + q0 + r + kh * 8;
      ctx[(size_t)m * 1024 + n] = f2bf(O[i][r] * inv[r]);
    }
  }
}

// ---------------------------------------------------------------------------
// In-place LayerNorm over rows of length 1024 (one 256-thread block per row).
// ---------------------------------------------------------------------------
__global__ __launch_bounds__(256) void layernorm_inplace(float* __restrict__ Y,
                                                         const float* __restrict__ w,
                                                         const float* __restrict__ b) {
  __shared__ float red1[8];
  __shared__ float red2[8];
  float* y = Y + (size_t)blockIdx.x * 1024;
  const int tid = threadIdx.x;
  float x[4];
  float s = 0.f;
#pragma unroll
  for (int i = 0; i < 4; i++) { x[i] = y[tid + i * 256]; s += x[i]; }
#pragma unroll
  for (int m = 1; m < 32; m <<= 1) s += __shfl_xor(s, m, 32);
  if ((tid & 31) == 0) red1[tid >> 5] = s;
  __syncthreads();
  float t1 = red1[tid & 7];
#pragma unroll
  for (int m = 1; m < 8; m <<= 1) t1 += __shfl_xor(t1, m, 32);
  const float mean = t1 * (1.f / 1024.f);
  float ss = 0.f;
#pragma unroll
  for (int i = 0; i < 4; i++) { const float d = x[i] - mean; ss += d * d; }
#pragma unroll
  for (int m = 1; m < 32; m <<= 1) ss += __shfl_xor(ss, m, 32);
  if ((tid & 31) == 0) red2[tid >> 5] = ss;
  __syncthreads();
  float t2 = red2[tid & 7];
#pragma unroll
  for (int m = 1; m < 8; m <<= 1) t2 += __shfl_xor(t2, m, 32);
  const float rstd = rsqrtf(t2 * (1.f / 1024.f) + 1e-12f);
#pragma unroll
  for (int i = 0; i < 4; i++) {
    const int c = tid + i * 256;
    y[c] = w[c] * (x[i] - mean) * rstd + b[c];
  }
}

// ---------------------------------------------------------------------------
extern "C" void kernel_launch(void* const* d_in, const int* in_sizes, int n_in,
                              void* d_out, int out_size, void* d_ws, size_t ws_size,
                              hipStream_t stream) {
  (void)in_sizes; (void)n_in; (void)out_size; (void)ws_size;
  const float* G    = (const float*)d_in[0];
  const float* T    = (const float*)d_in[1];
  const float* mask = (const float*)d_in[2];
  const float* Wq   = (const float*)d_in[3];
  const float* bq   = (const float*)d_in[4];
  const float* WqT  = (const float*)d_in[5];
  const float* bqT  = (const float*)d_in[6];
  const float* Wk   = (const float*)d_in[7];
  const float* bk   = (const float*)d_in[8];
  const float* WkT  = (const float*)d_in[9];
  const float* bkT  = (const float*)d_in[10];
  const float* Wv   = (const float*)d_in[11];
  const float* bv   = (const float*)d_in[12];
  const float* WvT  = (const float*)d_in[13];
  const float* bvT  = (const float*)d_in[14];
  const float* Wg   = (const float*)d_in[15];
  const float* bg   = (const float*)d_in[16];
  const float* gw   = (const float*)d_in[17];
  const float* gb   = (const float*)d_in[18];
  const float* Wt   = (const float*)d_in[19];
  const float* bt   = (const float*)d_in[20];
  const float* tw   = (const float*)d_in[21];
  const float* tb   = (const float*)d_in[22];

  char* ws = (char*)d_ws;
  size_t off = 0;
  auto take = [&](size_t bytes) -> char* {
    char* p = ws + off;
    off += (bytes + 255) & ~(size_t)255;
    return p;
  };
  const size_t TOK = (size_t)8192 * 1024;  // tokens * hidden

  unsigned short* Gb   = (unsigned short*)take(TOK * 2);
  unsigned short* Tb   = (unsigned short*)take(TOK * 2);
  unsigned short* Wqb  = (unsigned short*)take((size_t)2048 * 1024 * 2);
  unsigned short* WqTb = (unsigned short*)take((size_t)2048 * 1024 * 2);
  unsigned short* Wkb  = (unsigned short*)take((size_t)2048 * 1024 * 2);
  unsigned short* WkTb = (unsigned short*)take((size_t)2048 * 1024 * 2);
  unsigned short* Wvb  = (unsigned short*)take((size_t)1024 * 1024 * 2);
  unsigned short* WvTb = (unsigned short*)take((size_t)1024 * 1024 * 2);
  unsigned short* Wgb  = (unsigned short*)take((size_t)1024 * 1024 * 2);
  unsigned short* Wtb  = (unsigned short*)take((size_t)1024 * 1024 * 2);
  unsigned short* Qb   = (unsigned short*)take((size_t)8192 * 2048 * 2);
  unsigned short* Kbuf = (unsigned short*)take((size_t)8192 * 2048 * 2);
  unsigned short* QTb  = (unsigned short*)take((size_t)8192 * 2048 * 2);
  unsigned short* KTb  = (unsigned short*)take((size_t)8192 * 2048 * 2);
  unsigned short* VtA  = (unsigned short*)take(TOK * 2);  // [b,h,d,s] from T (branch G)
  unsigned short* VtB  = (unsigned short*)take(TOK * 2);  // [b,h,d,s] from G (branch T)
  unsigned short* ctxG = (unsigned short*)take(TOK * 2);
  unsigned short* ctxT = (unsigned short*)take(TOK * 2);

  auto cvt = [&](const float* src, unsigned short* dst, int n) {
    cvt_bf16<<<(n + 255) / 256, 256, 0, stream>>>(src, dst, n);
  };
  cvt(G, Gb, 8388608);
  cvt(T, Tb, 8388608);
  cvt(Wq, Wqb, 2097152);
  cvt(WqT, WqTb, 2097152);
  cvt(Wk, Wkb, 2097152);
  cvt(WkT, WkTb, 2097152);
  cvt(Wv, Wvb, 1048576);
  cvt(WvT, WvTb, 1048576);
  cvt(Wg, Wgb, 1048576);
  cvt(Wt, Wtb, 1048576);

  const dim3 blk(256);
  const dim3 g2048(8, 128), g1024(4, 128);  // block tile 64(M) x 256(N)

  // q = G*Wq^T + bq ; k = T*Wk^T + bk ; qT = T*WqT^T + bqT ; kT = G*WkT^T + bkT
  gemm_proj<0><<<g2048, blk, 0, stream>>>(Gb, Wqb, bq, nullptr, Qb, nullptr, 2048, 1024);
  gemm_proj<0><<<g2048, blk, 0, stream>>>(Tb, Wkb, bk, nullptr, Kbuf, nullptr, 2048, 1024);
  gemm_proj<0><<<g2048, blk, 0, stream>>>(Tb, WqTb, bqT, nullptr, QTb, nullptr, 2048, 1024);
  gemm_proj<0><<<g2048, blk, 0, stream>>>(Gb, WkTb, bkT, nullptr, KTb, nullptr, 2048, 1024);
  // v = T*Wv^T + bv (transposed per head) ; vT = G*WvT^T + bvT
  gemm_proj<1><<<g1024, blk, 0, stream>>>(Tb, Wvb, bv, nullptr, VtA, nullptr, 1024, 1024);
  gemm_proj<1><<<g1024, blk, 0, stream>>>(Gb, WvTb, bvT, nullptr, VtB, nullptr, 1024, 1024);

  // attention: ctx_G = softmax(q k^T) v ; ctx_T = softmax(qT kT^T) vT
  attn_fused<<<512, 128, 0, stream>>>(Qb, Kbuf, VtA, mask, ctxG);
  attn_fused<<<512, 128, 0, stream>>>(QTb, KTb, VtB, mask, ctxT);

  // SelfOutput dense + bias + residual (fp32 into d_out), then LayerNorm in place
  float* outG = (float*)d_out;
  float* outT = outG + TOK;
  gemm_proj<2><<<g1024, blk, 0, stream>>>(ctxG, Wgb, bg, G, nullptr, outG, 1024, 1024);
  gemm_proj<2><<<g1024, blk, 0, stream>>>(ctxT, Wtb, bt, T, nullptr, outT, 1024, 1024);

  layernorm_inplace<<<8192, 256, 0, stream>>>(outG, gw, gb);
  layernorm_inplace<<<8192, 256, 0, stream>>>(outT, tw, tb);
}
